// GIN_11450382811776
// MI455X (gfx1250) — matrix operand order repaired and checked by reading, not compiled, not run on hardware
//
#include <hip/hip_runtime.h>

// ---------------------------------------------------------------------------
// GIN over dense 16384x16384 adjacency on MI455X (gfx1250, wave32, WMMA).
// Dominant cost: 3x (A @ Z), A:16384^2, Z:16384x256  => 412 GFLOP.
// A converted to fp16 once; V_WMMA_F32_16X16X32_F16 with fp32 accumulation;
// Z kept transposed in fp16 so B fragments are single contiguous 32B loads.
// K loop: plain loop, unroll 2 — let the gfx1250 scheduler pipeline loads
// (it issues clauses + partial s_wait_loadcnt on its own); single fragment
// set keeps VGPR pressure low so accumulators stay pinned (no v_mov rotation).
// ---------------------------------------------------------------------------

typedef _Float16 v16h __attribute__((ext_vector_type(16)));
typedef _Float16 v8h  __attribute__((ext_vector_type(8)));
typedef float    v8f  __attribute__((ext_vector_type(8)));
typedef float    v4f  __attribute__((ext_vector_type(4)));

#define GN      16384   // graph nodes
#define IN_DIM  128
#define HID     128
#define ZDIM    256     // IN_DIM + HID
#define ODIM    64
#define NLAYERS 3

// ---------------------------------------------------------------------------
// 1) Convert A (fp32) -> A_h (fp16), 8 elements / thread.
// ---------------------------------------------------------------------------
__global__ void k_cvt_f32_f16(const float* __restrict__ src,
                              _Float16* __restrict__ dst) {
    long i = ((long)blockIdx.x * blockDim.x + threadIdx.x) * 8;
    const v4f* p = (const v4f*)(src + i);
    v4f a = p[0];
    v4f b = p[1];
    v8h o;
    o[0] = (_Float16)a[0]; o[1] = (_Float16)a[1];
    o[2] = (_Float16)a[2]; o[3] = (_Float16)a[3];
    o[4] = (_Float16)b[0]; o[5] = (_Float16)b[1];
    o[6] = (_Float16)b[2]; o[7] = (_Float16)b[3];
    *(v8h*)(dst + i) = o;
}

// ---------------------------------------------------------------------------
// 2) H0 = X @ in_W + in_b   (N x 128 x 128, trivial bandwidth)
// ---------------------------------------------------------------------------
__global__ void k_in_proj(const float* __restrict__ X,
                          const float* __restrict__ W,
                          const float* __restrict__ b,
                          float* __restrict__ H) {
    int id = blockIdx.x * blockDim.x + threadIdx.x;
    int r = id >> 7, c = id & 127;
    const float* x = X + (long)r * IN_DIM;
    float acc = b[c];
#pragma unroll 8
    for (int k = 0; k < IN_DIM; ++k)
        acc += x[k] * W[k * HID + c];
    H[id] = acc;
}

// ---------------------------------------------------------------------------
// 3) Zt[c][r] = fp16( c<128 ? X[r][c] : H[r][c-128] )   (LDS tile transpose)
//    grid (GN/32, ZDIM/32), block (32,8)
// ---------------------------------------------------------------------------
__global__ void k_build_zt(const float* __restrict__ X,
                           const float* __restrict__ H,
                           _Float16* __restrict__ Zt) {
    __shared__ _Float16 t[32][33];
    const int rt = blockIdx.x * 32, ct = blockIdx.y * 32;
    const int tx = threadIdx.x, ty = threadIdx.y;
#pragma unroll
    for (int i = 0; i < 4; ++i) {
        int rl = ty + i * 8;
        int r = rt + rl, c = ct + tx;
        float v = (c < IN_DIM) ? X[(long)r * IN_DIM + c]
                               : H[(long)r * HID + (c - IN_DIM)];
        t[rl][tx] = (_Float16)v;
    }
    __syncthreads();
#pragma unroll
    for (int i = 0; i < 4; ++i) {
        int cl = ty + i * 8;
        Zt[(long)(ct + cl) * GN + (rt + tx)] = t[tx][cl];
    }
}

// ---------------------------------------------------------------------------
// 4) U = A_h @ Z + Z   (the 412-GFLOP kernel)
//    block = 256 threads (8 waves) -> 128x128 tile of U
//    wave  = 32x64 tile = 2(m) x 4(n) WMMA accumulators.
// ---------------------------------------------------------------------------
__global__ __launch_bounds__(256) void k_gemm_az(
    const _Float16* __restrict__ Ah, const _Float16* __restrict__ Zt,
    const float* __restrict__ X, const float* __restrict__ Hin,
    float* __restrict__ U) {

    const int lane = threadIdx.x & 31;
    const int wid  = threadIdx.x >> 5;
    const int ll   = lane & 15;        // lane 0..15 id
    const int lh   = lane >> 4;        // lane half (K split)
    const int wm   = wid >> 1;         // 0..3  (32-row strips)
    const int wn   = wid & 1;          // 0..1  (64-col strips)

    const long row0 = (long)blockIdx.x * 128 + wm * 32;
    const int  col0 = blockIdx.y * 128 + wn * 64;

    // A fragment base pointers: per-lane row, per-mt +16 rows.
    const _Float16* aR0 = Ah + (row0 + ll) * (long)GN;
    const _Float16* aR1 = Ah + (row0 + 16 + ll) * (long)GN;
    // B fragment base pointers: per-lane column (= row of Zt).
    const _Float16* bR0 = Zt + (long)(col0 + ll) * GN;
    const _Float16* bR1 = Zt + (long)(col0 + 16 + ll) * GN;
    const _Float16* bR2 = Zt + (long)(col0 + 32 + ll) * GN;
    const _Float16* bR3 = Zt + (long)(col0 + 48 + ll) * GN;

    // 16x32 f16 A operand: lanes 0-15 rows 0-15; VGPR0-7 = K[lh*16 .. +15]
    //  -> one contiguous 32B load.
    auto loadA = [&](const _Float16* base, int kb) -> v16h {
        return __builtin_nontemporal_load((const v16h*)(base + kb + lh * 16));
    };
    // 32x16 f16 B operand: lanes 0-15 hold K=0-15 of their column (packed
    // 2/VGPR), lanes 16-31 hold K=16-31 -> one contiguous 32B load from Zt.
    auto loadB = [&](const _Float16* base, int kb) -> v16h {
        return *(const v16h*)(base + kb + lh * 16);
    };

    v8f acc00 = (v8f)(0.0f), acc01 = (v8f)(0.0f),
        acc02 = (v8f)(0.0f), acc03 = (v8f)(0.0f),
        acc10 = (v8f)(0.0f), acc11 = (v8f)(0.0f),
        acc12 = (v8f)(0.0f), acc13 = (v8f)(0.0f);

    // Plain K loop: 512 slabs of depth 32. The gfx1250 backend hoists the
    // (independent) loads of the next unrolled iteration above the WMMAs of
    // the current one and inserts partial s_wait_loadcnt on its own.
#pragma unroll 2
    for (int kb = 0; kb < GN; kb += 32) {
        v16h a0 = loadA(aR0, kb);
        v16h a1 = loadA(aR1, kb);
        v16h b0 = loadB(bR0, kb);
        v16h b1 = loadB(bR1, kb);
        v16h b2 = loadB(bR2, kb);
        v16h b3 = loadB(bR3, kb);
        acc00 = __builtin_amdgcn_wmma_f32_16x16x32_f16(false, a0, false, b0, (short)0, acc00, false, false);
        acc01 = __builtin_amdgcn_wmma_f32_16x16x32_f16(false, a0, false, b1, (short)0, acc01, false, false);
        acc02 = __builtin_amdgcn_wmma_f32_16x16x32_f16(false, a0, false, b2, (short)0, acc02, false, false);
        acc03 = __builtin_amdgcn_wmma_f32_16x16x32_f16(false, a0, false, b3, (short)0, acc03, false, false);
        acc10 = __builtin_amdgcn_wmma_f32_16x16x32_f16(false, a1, false, b0, (short)0, acc10, false, false);
        acc11 = __builtin_amdgcn_wmma_f32_16x16x32_f16(false, a1, false, b1, (short)0, acc11, false, false);
        acc12 = __builtin_amdgcn_wmma_f32_16x16x32_f16(false, a1, false, b2, (short)0, acc12, false, false);
        acc13 = __builtin_amdgcn_wmma_f32_16x16x32_f16(false, a1, false, b3, (short)0, acc13, false, false);
    }

    // Epilogue: U = acc + Z (residual), Z recomputed from fp32 X/H.
    // C/D layout: VGPR j -> row (j + 8*lh), col = lane&15 within 16x16 tile.
    const v8f* accs[2][4] = { { &acc00, &acc01, &acc02, &acc03 },
                              { &acc10, &acc11, &acc12, &acc13 } };
#pragma unroll
    for (int mt = 0; mt < 2; ++mt) {
#pragma unroll
        for (int nt = 0; nt < 4; ++nt) {
            const v8f a = *accs[mt][nt];
            const long rb = row0 + mt * 16 + lh * 8;
            const int  c  = col0 + nt * 16 + ll;
#pragma unroll
            for (int j = 0; j < 8; ++j) {
                const long r = rb + j;
                float zv = (c < IN_DIM) ? X[r * IN_DIM + c]
                                        : Hin[r * HID + (c - IN_DIM)];
                U[r * ZDIM + c] = a[j] + zv;
            }
        }
    }
}

// ---------------------------------------------------------------------------
// 5) H' = relu(U @ conv_W[l] + conv_b[l])   (N x 256 x 128)
// ---------------------------------------------------------------------------
__global__ void k_conv_relu(const float* __restrict__ U,
                            const float* __restrict__ W,
                            const float* __restrict__ b,
                            float* __restrict__ Hn) {
    int id = blockIdx.x * blockDim.x + threadIdx.x;
    int r = id >> 7, c = id & 127;
    const float* u = U + (long)r * ZDIM;
    float acc = b[c];
#pragma unroll 8
    for (int k = 0; k < ZDIM; ++k)
        acc += u[k] * W[k * HID + c];
    Hn[id] = acc > 0.0f ? acc : 0.0f;
}

// ---------------------------------------------------------------------------
// 6) out = [H0|H1|H2|H3] @ out_W + out_b   (N x 512 x 64)
// ---------------------------------------------------------------------------
__global__ void k_out_proj(const float* __restrict__ H0,
                           const float* __restrict__ H1,
                           const float* __restrict__ H2,
                           const float* __restrict__ H3,
                           const float* __restrict__ W,
                           const float* __restrict__ b,
                           float* __restrict__ out) {
    int id = blockIdx.x * blockDim.x + threadIdx.x;
    int r = id >> 6, o = id & 63;
    const float* hs[4] = { H0 + (long)r * HID, H1 + (long)r * HID,
                           H2 + (long)r * HID, H3 + (long)r * HID };
    float acc = b[o];
#pragma unroll
    for (int l = 0; l < 4; ++l) {
        const float* h = hs[l];
        const float* w = W + (long)l * HID * ODIM + o;
#pragma unroll 8
        for (int j = 0; j < HID; ++j)
            acc += h[j] * w[j * ODIM];
    }
    out[id] = acc;
}

// ---------------------------------------------------------------------------
extern "C" void kernel_launch(void* const* d_in, const int* in_sizes, int n_in,
                              void* d_out, int out_size, void* d_ws, size_t ws_size,
                              hipStream_t stream) {
    const float* A      = (const float*)d_in[0];
    const float* X      = (const float*)d_in[1];
    const float* in_W   = (const float*)d_in[2];
    const float* in_b   = (const float*)d_in[3];
    const float* conv_W = (const float*)d_in[4];
    const float* conv_b = (const float*)d_in[5];
    const float* out_W  = (const float*)d_in[6];
    const float* out_b  = (const float*)d_in[7];
    float* out = (float*)d_out;

    // workspace carve-out (256B aligned)
    char* ws = (char*)d_ws;
    size_t off = 0;
    auto carve = [&](size_t bytes) -> void* {
        void* p = ws + off;
        off = (off + bytes + 255) & ~(size_t)255;
        return p;
    };
    _Float16* Ah = (_Float16*)carve((size_t)GN * GN * sizeof(_Float16)); // 512 MB
    _Float16* Zt = (_Float16*)carve((size_t)ZDIM * GN * sizeof(_Float16));
    float*    U  = (float*)   carve((size_t)GN * ZDIM * sizeof(float));
    float*    H[NLAYERS + 1];
    for (int i = 0; i <= NLAYERS; ++i)
        H[i] = (float*)carve((size_t)GN * HID * sizeof(float));

    // A fp32 -> fp16 (once)
    k_cvt_f32_f16<<<(unsigned)((size_t)GN * GN / 8 / 256), 256, 0, stream>>>(A, Ah);
    // H0 = X @ in_W + in_b
    k_in_proj<<<GN * HID / 256, 256, 0, stream>>>(X, in_W, in_b, H[0]);

    for (int l = 0; l < NLAYERS; ++l) {
        k_build_zt<<<dim3(GN / 32, ZDIM / 32), dim3(32, 8), 0, stream>>>(X, H[l], Zt);
        k_gemm_az<<<dim3(GN / 128, ZDIM / 128), 256, 0, stream>>>(Ah, Zt, X, H[l], U);
        k_conv_relu<<<GN * HID / 256, 256, 0, stream>>>(
            U, conv_W + (size_t)l * ZDIM * HID, conv_b + (size_t)l * HID, H[l + 1]);
    }

    k_out_proj<<<GN * ODIM / 256, 256, 0, stream>>>(
        H[0], H[1], H[2], H[3], out_W, out_b, out);
}
